// Mamba2DBlock_52845277610276
// MI455X (gfx1250) — matrix operand order repaired and checked by reading, not compiled, not run on hardware
//
#include <hip/hip_runtime.h>
#include <hip/hip_bf16.h>

typedef __attribute__((ext_vector_type(16))) _Float16 v16h;
typedef __attribute__((ext_vector_type(8)))  _Float16 v8h;
typedef __attribute__((ext_vector_type(8)))  float    v8f;

// ---------------- problem constants ----------------
constexpr int Bn  = 8;
constexpr int Cn  = 192;
constexpr int Hn  = 128;
constexpr int Wn  = 128;
constexpr int HWn = Hn * Wn;                      // 16384
constexpr long long Mn = (long long)Bn * HWn;     // 131072 pixels
constexpr int C2  = 384;                          // 2*C
constexpr int HIDn = 384;
constexpr float EPSc = 1e-5f;

// ---------------- workspace layout (bytes) ----------------
constexpr size_t OFF_SUM1 = 0;     // 8 doubles
constexpr size_t OFF_SQ1  = 64;
constexpr size_t OFF_SUM2 = 128;
constexpr size_t OFF_SQ2  = 192;
constexpr size_t OFF_MU1  = 256;   // 8 floats each
constexpr size_t OFF_RS1  = 288;
constexpr size_t OFF_MU2  = 320;
constexpr size_t OFF_RS2  = 352;
constexpr size_t OFF_WIN  = 1024;                         // f16 [384][192]  ([N][K])
constexpr size_t OFF_WOUT = OFF_WIN  + (size_t)C2*Cn*2;   // f16 [192][192]
constexpr size_t OFF_W1   = OFF_WOUT + (size_t)Cn*Cn*2;   // f16 [384][192]
constexpr size_t OFF_W2   = OFF_W1   + (size_t)HIDn*Cn*2; // f16 [192][384]
constexpr size_t SZ_HC = (size_t)Mn * Cn * 2;   // 50331648  (f16 M x C)
constexpr size_t SZ_FC = (size_t)Mn * Cn * 4;   // 100663296 (f32 M x C)
constexpr size_t OFF_XH    = 524288;            // f16 xn NHWC (later: mh)
constexpr size_t OFF_U     = OFF_XH    + SZ_HC; // f32 u NHWC (later: col tmp, then h1 f16 MxHID)
constexpr size_t OFF_GATE  = OFF_U     + SZ_FC; // f16 gate NHWC
constexpr size_t OFF_UDW   = OFF_GATE  + SZ_HC; // f32 dwconv out (later: y2)
constexpr size_t OFF_USCAN = OFF_UDW   + SZ_FC; // f32 scan accum (later: m2)
constexpr size_t OFF_UH    = OFF_USCAN + SZ_FC; // f16 gated scan out
constexpr size_t OFF_X2    = OFF_UH    + SZ_HC; // f32 residual NHWC

__device__ __forceinline__ float sigmoidf_(float x) { return 1.0f / (1.0f + __expf(-x)); }

// ---------------- tiny helpers ----------------
__global__ void k_zero(double* p, int n) {
    int i = threadIdx.x;
    if (i < n) p[i] = 0.0;
}

// f32 -> f16 cast, layout preserved ([O][C] == [N][K], which is what the
// B-fragment wants: lane = N column, 16 contiguous K halves per fragment)
__global__ void k_wcast(const float* __restrict__ src, _Float16* __restrict__ dst, int n) {
    int i = blockIdx.x * blockDim.x + threadIdx.x;
    if (i < n) dst[i] = (_Float16)src[i];
}

__global__ void k_finalize(const double* __restrict__ sum, const double* __restrict__ sq,
                           float* __restrict__ mu, float* __restrict__ rs, double invN) {
    int b = threadIdx.x;
    if (b < Bn) {
        double m = sum[b] * invN;
        double v = sq[b] * invN - m * m;
        mu[b] = (float)m;
        rs[b] = rsqrtf((float)v + EPSc);
    }
}

// ---------------- group-norm 1 statistics over x (NCHW f32) ----------------
constexpr int BPB = 96;  // blocks per batch
__global__ __launch_bounds__(256) void k_gn1_stats(const float* __restrict__ x,
                                                   double* __restrict__ sum,
                                                   double* __restrict__ sq) {
    const int b   = blockIdx.x / BPB;
    const int seg = blockIdx.x % BPB;
    const size_t Nper = (size_t)Cn * HWn;                 // 3145728
    const size_t per_block = Nper / BPB;                  // 32768
    const float4* base = (const float4*)(x + (size_t)b * Nper + (size_t)seg * per_block);
    float s = 0.f, ss = 0.f;
    for (int it = 0; it < 32; ++it) {
        float4 v = base[it * 256 + threadIdx.x];
        s  += v.x + v.y + v.z + v.w;
        ss += v.x * v.x + v.y * v.y + v.z * v.z + v.w * v.w;
    }
    __shared__ float sh_s[256], sh_q[256];
    sh_s[threadIdx.x] = s; sh_q[threadIdx.x] = ss;
    __syncthreads();
    for (int o = 128; o > 0; o >>= 1) {
        if (threadIdx.x < o) {
            sh_s[threadIdx.x] += sh_s[threadIdx.x + o];
            sh_q[threadIdx.x] += sh_q[threadIdx.x + o];
        }
        __syncthreads();
    }
    if (threadIdx.x == 0) {
        atomicAdd(&sum[b], (double)sh_s[0]);
        atomicAdd(&sq[b],  (double)sh_q[0]);
    }
}

// ---------------- GN1 apply: x NCHW f32 -> xh NHWC f16 (tiled transpose) ----------------
__global__ __launch_bounds__(256) void k_gn1_apply(const float* __restrict__ x,
                                                   const float* __restrict__ gw,
                                                   const float* __restrict__ gb,
                                                   const float* __restrict__ mu,
                                                   const float* __restrict__ rs,
                                                   _Float16* __restrict__ xh) {
    const int b = blockIdx.z / Hn, h = blockIdx.z % Hn;
    const int c0 = blockIdx.y * 32, w0 = blockIdx.x * 32;
    __shared__ float t[32][33];
    const int tx = threadIdx.x, ty = threadIdx.y;
#pragma unroll
    for (int i = 0; i < 4; ++i) {
        int c = c0 + ty + i * 8;
        t[ty + i * 8][tx] = x[(((size_t)b * Cn + c) * Hn + h) * Wn + w0 + tx];
    }
    __syncthreads();
    const float m = mu[b], r = rs[b];
#pragma unroll
    for (int i = 0; i < 4; ++i) {
        int w = w0 + ty + i * 8;
        int c = c0 + tx;
        float v = (t[tx][ty + i * 8] - m) * r * gw[c] + gb[c];
        xh[(((size_t)b * Hn + h) * Wn + w) * Cn + c] = (_Float16)v;
    }
}

// ---------------- WMMA GEMM: D[M][N] = A[M][K](f16) * W[N][K](f16)^T + bias ----------------
// Each wave: NT n-tiles of 16, 4 m-tiles of 16, all K/32 B-fragments preloaded.
// MODE 0: split -> outF = u (f32, ld 192) for n<192, outH = gate (f16, ld 192) for n>=192
// MODE 1: outF f32 [M][N]
// MODE 2: GELU -> outH f16 [M][N]
template <int KT, int NT, int MODE>
__global__ __launch_bounds__(128) void k_gemm(const _Float16* __restrict__ A,
                                              const _Float16* __restrict__ Wt,
                                              const float* __restrict__ bias,
                                              float* __restrict__ outF,
                                              _Float16* __restrict__ outH,
                                              int N) {
    constexpr int K = KT * 32;
    const int lane = threadIdx.x & 31;
    const int wave = threadIdx.x >> 5;
    const int n0 = blockIdx.x * (16 * NT);
    const int m0 = (blockIdx.y * 4 + wave) * 64;   // 4 m-tiles of 16 per wave
    const int lane15 = lane & 15;

    // preload all B fragments: lane = N column, 16 contiguous K halves (32B load)
    v16h bf[NT][KT];
    const int kg = (lane >> 4) * 16;
    float bv[NT];
#pragma unroll
    for (int nt = 0; nt < NT; ++nt) {
        const int n = n0 + nt * 16 + lane15;
        const _Float16* p = Wt + (size_t)n * K + kg;
#pragma unroll
        for (int t = 0; t < KT; ++t) bf[nt][t] = *(const v16h*)(p + t * 32);
        bv[nt] = bias[n];
    }
    const int mrow = lane15;
    const int kb   = (lane >> 4) * 16;
    const int mofs = (lane >> 4) << 3;

#pragma unroll
    for (int mt = 0; mt < 4; ++mt) {
        const int mbase = m0 + mt * 16;
        v8f acc[NT];
#pragma unroll
        for (int nt = 0; nt < NT; ++nt)
#pragma unroll
            for (int e = 0; e < 8; ++e) acc[nt][e] = 0.0f;
        const _Float16* arow = A + (size_t)(mbase + mrow) * K + kb;
#pragma unroll
        for (int t = 0; t < KT; ++t) {
            v8h lo = *(const v8h*)(arow + t * 32);
            v8h hi = *(const v8h*)(arow + t * 32 + 8);
            v16h af = __builtin_shufflevector(lo, hi, 0, 1, 2, 3, 4, 5, 6, 7,
                                              8, 9, 10, 11, 12, 13, 14, 15);
#pragma unroll
            for (int nt = 0; nt < NT; ++nt)
                acc[nt] = __builtin_amdgcn_wmma_f32_16x16x32_f16(
                    false, af, false, bf[nt][t], (short)0, acc[nt], false, false);
        }
        // epilogue: D layout -> VGPR r: rows mbase+r (lanes 0-15) / mbase+r+8 (16-31)
        const size_t mrow0 = (size_t)(mbase + mofs);
#pragma unroll
        for (int nt = 0; nt < NT; ++nt) {
            const int col = n0 + nt * 16 + lane15;
            if (MODE == 0) {
                if (n0 + nt * 16 < 192) {           // wave-uniform per unrolled nt
                    float* po = outF + mrow0 * 192 + col;
#pragma unroll
                    for (int r = 0; r < 8; ++r) po[(size_t)r * 192] = acc[nt][r] + bv[nt];
                } else {
                    _Float16* ph = outH + mrow0 * 192 + (col - 192);
#pragma unroll
                    for (int r = 0; r < 8; ++r)
                        ph[(size_t)r * 192] = (_Float16)(acc[nt][r] + bv[nt]);
                }
            } else if (MODE == 1) {
                float* po = outF + mrow0 * N + col;
#pragma unroll
                for (int r = 0; r < 8; ++r) po[(size_t)r * N] = acc[nt][r] + bv[nt];
            } else {
                _Float16* ph = outH + mrow0 * N + col;
#pragma unroll
                for (int r = 0; r < 8; ++r) {
                    float v = acc[nt][r] + bv[nt];
                    ph[(size_t)r * N] =
                        (_Float16)(0.5f * v * (1.0f + erff(v * 0.70710678118f)));
                }
            }
        }
    }
}

// ---------------- depthwise 3x3 'SAME' on u (NHWC f32) ----------------
__global__ __launch_bounds__(256) void k_dw3x3(const float* __restrict__ u,
                                               const float* __restrict__ dww,
                                               const float* __restrict__ dwb,
                                               float* __restrict__ udw) {
    size_t idx = (size_t)blockIdx.x * blockDim.x + threadIdx.x;
    if (idx >= (size_t)Mn * Cn) return;
    const int c = (int)(idx % Cn);
    const size_t p = idx / Cn;
    const int w = (int)(p % Wn);
    const int h = (int)((p / Wn) % Hn);
    const size_t bh = p / Wn;   // b*Hn + h
    float s = dwb[c];
#pragma unroll
    for (int kh = 0; kh < 3; ++kh) {
        int h2 = h + kh - 1;
        if (h2 < 0 || h2 >= Hn) continue;
#pragma unroll
        for (int kw = 0; kw < 3; ++kw) {
            int w2 = w + kw - 1;
            if (w2 < 0 || w2 >= Wn) continue;
            s += u[(((bh - h) + h2) * (size_t)Wn + w2) * Cn + c] * dww[c * 9 + kh * 3 + kw];
        }
    }
    udw[idx] = s;
}

// ---------------- bidirectional row scan (along W): udw -> uscan ----------------
__global__ __launch_bounds__(192) void k_scan_rows(const float* __restrict__ udw,
                                                   const float* __restrict__ al,
                                                   const float* __restrict__ be,
                                                   const float* __restrict__ ga,
                                                   const float* __restrict__ de,
                                                   float* __restrict__ uscan) {
    const int c = threadIdx.x;
    const float a = sigmoidf_(al[c]), B_ = be[c], G = ga[c], D = de[c];
    const size_t base = (size_t)blockIdx.x * Wn * Cn + c;   // (b,h) sequence
    float s = 0.f;
    for (int w = 0; w < Wn; ++w) {
        float x = udw[base + (size_t)w * Cn];
        s = s * a + x * B_;
        uscan[base + (size_t)w * Cn] = s * G + x * D;        // yf
    }
    s = 0.f;
    for (int w = Wn - 1; w >= 0; --w) {
        float x = udw[base + (size_t)w * Cn];
        s = s * a + x * B_;
        float yb = s * G + x * D;
        uscan[base + (size_t)w * Cn] = 0.5f * (uscan[base + (size_t)w * Cn] + yb);
    }
}

// ---------------- bidirectional col scan (along H) + merge + gate -> uh f16 ----------------
__global__ __launch_bounds__(192) void k_scan_cols(const float* __restrict__ udw,
                                                   const float* __restrict__ al,
                                                   const float* __restrict__ be,
                                                   const float* __restrict__ ga,
                                                   const float* __restrict__ de,
                                                   const float* __restrict__ uscan,
                                                   const _Float16* __restrict__ gate,
                                                   float* __restrict__ tmp,
                                                   _Float16* __restrict__ uh) {
    const int c = threadIdx.x;
    const int b = blockIdx.x / Wn, w = blockIdx.x % Wn;
    const float a = sigmoidf_(al[c]), B_ = be[c], G = ga[c], D = de[c];
    const size_t base = (((size_t)b * Hn) * Wn + w) * Cn + c;
    const size_t stride = (size_t)Wn * Cn;
    float s = 0.f;
    for (int h = 0; h < Hn; ++h) {
        float x = udw[base + (size_t)h * stride];
        s = s * a + x * B_;
        tmp[base + (size_t)h * stride] = s * G + x * D;       // yf
    }
    s = 0.f;
    for (int h = Hn - 1; h >= 0; --h) {
        size_t i = base + (size_t)h * stride;
        float x = udw[i];
        s = s * a + x * B_;
        float ycol = 0.5f * (tmp[i] + (s * G + x * D));
        float v = 0.5f * (uscan[i] + ycol);
        float g = (float)gate[i];
        uh[i] = (_Float16)(v * sigmoidf_(g));
    }
}

// ---------------- residual add (y2 NHWC + x NCHW) -> x2 NHWC, fused GN2 stats ----------------
__global__ __launch_bounds__(256) void k_residual_gn2(const float* __restrict__ y2,
                                                      const float* __restrict__ x,
                                                      float* __restrict__ x2,
                                                      double* __restrict__ sum,
                                                      double* __restrict__ sq) {
    const int b = blockIdx.z / Hn, h = blockIdx.z % Hn;
    const int c0 = blockIdx.y * 32, w0 = blockIdx.x * 32;
    __shared__ float t[32][33];
    const int tx = threadIdx.x, ty = threadIdx.y;
#pragma unroll
    for (int i = 0; i < 4; ++i) {
        int c = c0 + ty + i * 8;
        t[ty + i * 8][tx] = x[(((size_t)b * Cn + c) * Hn + h) * Wn + w0 + tx];
    }
    __syncthreads();
    float s = 0.f, ss = 0.f;
#pragma unroll
    for (int i = 0; i < 4; ++i) {
        int w = w0 + ty + i * 8;
        int c = c0 + tx;
        size_t idx = (((size_t)b * Hn + h) * Wn + w) * Cn + c;
        float v = y2[idx] + t[tx][ty + i * 8];
        x2[idx] = v;
        s += v; ss += v * v;
    }
    __shared__ float sh_s[256], sh_q[256];
    int tid = ty * 32 + tx;
    sh_s[tid] = s; sh_q[tid] = ss;
    __syncthreads();
    for (int o = 128; o > 0; o >>= 1) {
        if (tid < o) { sh_s[tid] += sh_s[tid + o]; sh_q[tid] += sh_q[tid + o]; }
        __syncthreads();
    }
    if (tid == 0) {
        atomicAdd(&sum[b], (double)sh_s[0]);
        atomicAdd(&sq[b],  (double)sh_q[0]);
    }
}

// ---------------- GN2 apply: x2 NHWC f32 -> mh NHWC f16 (elementwise) ----------------
__global__ __launch_bounds__(256) void k_gn2_apply(const float* __restrict__ x2,
                                                   const float* __restrict__ gw,
                                                   const float* __restrict__ gb,
                                                   const float* __restrict__ mu,
                                                   const float* __restrict__ rs,
                                                   _Float16* __restrict__ mh) {
    size_t idx = (size_t)blockIdx.x * blockDim.x + threadIdx.x;
    if (idx >= (size_t)Mn * Cn) return;
    const int c = (int)(idx % Cn);
    const int b = (int)(idx / ((size_t)Cn * HWn));
    float v = (x2[idx] - mu[b]) * rs[b] * gw[c] + gb[c];
    mh[idx] = (_Float16)v;
}

// ---------------- final: out NCHW f32 = x2 + m2 (NHWC), tiled transpose ----------------
__global__ __launch_bounds__(256) void k_final(const float* __restrict__ x2,
                                               const float* __restrict__ m2,
                                               float* __restrict__ out) {
    const int b = blockIdx.z / Hn, h = blockIdx.z % Hn;
    const int c0 = blockIdx.y * 32, w0 = blockIdx.x * 32;
    __shared__ float t[32][33];
    const int tx = threadIdx.x, ty = threadIdx.y;
#pragma unroll
    for (int i = 0; i < 4; ++i) {
        int w = w0 + ty + i * 8;
        int c = c0 + tx;
        size_t idx = (((size_t)b * Hn + h) * Wn + w) * Cn + c;
        t[tx][ty + i * 8] = x2[idx] + m2[idx];
    }
    __syncthreads();
#pragma unroll
    for (int i = 0; i < 4; ++i) {
        int c = c0 + ty + i * 8;
        out[(((size_t)b * Cn + c) * Hn + h) * Wn + w0 + tx] = t[ty + i * 8][tx];
    }
}

// ---------------- host launch ----------------
extern "C" void kernel_launch(void* const* d_in, const int* in_sizes, int n_in,
                              void* d_out, int out_size, void* d_ws, size_t ws_size,
                              hipStream_t stream) {
    (void)in_sizes; (void)n_in; (void)out_size; (void)ws_size;
    const float* x         = (const float*)d_in[0];
    const float* norm_w    = (const float*)d_in[1];
    const float* norm_b    = (const float*)d_in[2];
    const float* inproj_w  = (const float*)d_in[3];
    const float* inproj_b  = (const float*)d_in[4];
    const float* dw_w      = (const float*)d_in[5];
    const float* dw_b      = (const float*)d_in[6];
    const float* sh_alpha  = (const float*)d_in[7];
    const float* sh_beta   = (const float*)d_in[8];
    const float* sh_gamma  = (const float*)d_in[9];
    const float* sh_delta  = (const float*)d_in[10];
    const float* sv_alpha  = (const float*)d_in[11];
    const float* sv_beta   = (const float*)d_in[12];
    const float* sv_gamma  = (const float*)d_in[13];
    const float* sv_delta  = (const float*)d_in[14];
    const float* outproj_w = (const float*)d_in[15];
    const float* outproj_b = (const float*)d_in[16];
    const float* mlp_nw    = (const float*)d_in[17];
    const float* mlp_nb    = (const float*)d_in[18];
    const float* mlp_w1    = (const float*)d_in[19];
    const float* mlp_b1    = (const float*)d_in[20];
    const float* mlp_w2    = (const float*)d_in[21];
    const float* mlp_b2    = (const float*)d_in[22];

    char* wsb = (char*)d_ws;
    double* sum1 = (double*)(wsb + OFF_SUM1);
    double* sq1  = (double*)(wsb + OFF_SQ1);
    double* sum2 = (double*)(wsb + OFF_SUM2);
    double* sq2  = (double*)(wsb + OFF_SQ2);
    float* mu1 = (float*)(wsb + OFF_MU1);
    float* rs1 = (float*)(wsb + OFF_RS1);
    float* mu2 = (float*)(wsb + OFF_MU2);
    float* rs2 = (float*)(wsb + OFF_RS2);
    _Float16* w_in  = (_Float16*)(wsb + OFF_WIN);
    _Float16* w_out = (_Float16*)(wsb + OFF_WOUT);
    _Float16* w_m1  = (_Float16*)(wsb + OFF_W1);
    _Float16* w_m2  = (_Float16*)(wsb + OFF_W2);
    _Float16* xh    = (_Float16*)(wsb + OFF_XH);    // also mh
    float*    u     = (float*)(wsb + OFF_U);        // also col tmp
    _Float16* h1    = (_Float16*)(wsb + OFF_U);     // reuse for mlp hidden
    _Float16* gate  = (_Float16*)(wsb + OFF_GATE);
    float*    udw   = (float*)(wsb + OFF_UDW);
    float*    y2    = (float*)(wsb + OFF_UDW);      // reuse after scans
    float*    uscan = (float*)(wsb + OFF_USCAN);
    float*    m2    = (float*)(wsb + OFF_USCAN);    // reuse after GEMM4
    _Float16* uh    = (_Float16*)(wsb + OFF_UH);
    float*    x2    = (float*)(wsb + OFF_X2);

    const double invN = 1.0 / ((double)Cn * HWn);
    const size_t MC = (size_t)Mn * Cn;

    k_zero<<<1, 64, 0, stream>>>((double*)wsb, 32);

    k_wcast<<<(C2 * Cn + 255) / 256, 256, 0, stream>>>(inproj_w, w_in, C2 * Cn);
    k_wcast<<<(Cn * Cn + 255) / 256, 256, 0, stream>>>(outproj_w, w_out, Cn * Cn);
    k_wcast<<<(HIDn * Cn + 255) / 256, 256, 0, stream>>>(mlp_w1, w_m1, HIDn * Cn);
    k_wcast<<<(Cn * HIDn + 255) / 256, 256, 0, stream>>>(mlp_w2, w_m2, Cn * HIDn);

    k_gn1_stats<<<Bn * BPB, 256, 0, stream>>>(x, sum1, sq1);
    k_finalize<<<1, 8, 0, stream>>>(sum1, sq1, mu1, rs1, invN);
    k_gn1_apply<<<dim3(Wn / 32, Cn / 32, Bn * Hn), dim3(32, 8), 0, stream>>>(
        x, norm_w, norm_b, mu1, rs1, xh);

    // inproj GEMM: M x 192 x 384 -> u (f32) + gate (f16)
    k_gemm<6, 2, 0><<<dim3(C2 / 32, (int)(Mn / 256)), 128, 0, stream>>>(
        xh, w_in, inproj_b, u, gate, C2);

    k_dw3x3<<<(int)((MC + 255) / 256), 256, 0, stream>>>(u, dw_w, dw_b, udw);

    k_scan_rows<<<Bn * Hn, Cn, 0, stream>>>(udw, sh_alpha, sh_beta, sh_gamma, sh_delta, uscan);
    k_scan_cols<<<Bn * Wn, Cn, 0, stream>>>(udw, sv_alpha, sv_beta, sv_gamma, sv_delta,
                                            uscan, gate, u, uh);

    // outproj GEMM: M x 192 x 192 -> y2 (f32)
    k_gemm<6, 2, 1><<<dim3(Cn / 32, (int)(Mn / 256)), 128, 0, stream>>>(
        uh, w_out, outproj_b, y2, nullptr, Cn);

    k_residual_gn2<<<dim3(Wn / 32, Cn / 32, Bn * Hn), dim3(32, 8), 0, stream>>>(
        y2, x, x2, sum2, sq2);
    k_finalize<<<1, 8, 0, stream>>>(sum2, sq2, mu2, rs2, invN);
    k_gn2_apply<<<(int)((MC + 255) / 256), 256, 0, stream>>>(x2, mlp_nw, mlp_nb, mu2, rs2, xh);

    // mlp1 GEMM + GELU: M x 192 x 384 -> h1 (f16)
    k_gemm<6, 2, 2><<<dim3(HIDn / 32, (int)(Mn / 256)), 128, 0, stream>>>(
        xh, w_m1, mlp_b1, nullptr, h1, HIDn);

    // mlp2 GEMM: M x 384 x 192 -> m2 (f32)
    k_gemm<12, 1, 1><<<dim3(Cn / 16, (int)(Mn / 256)), 128, 0, stream>>>(
        h1, w_m2, mlp_b2, m2, nullptr, Cn);

    k_final<<<dim3(Wn / 32, Cn / 32, Bn * Hn), dim3(32, 8), 0, stream>>>(x2, m2, (float*)d_out);
}
